// TransformerDecoder_1417339208227
// MI455X (gfx1250) — matrix operand order repaired and checked by reading, not compiled, not run on hardware
//
#include <hip/hip_runtime.h>
#include <stdint.h>

// ---------------------------------------------------------------------------
// Types for CDNA5 WMMA (wave32): D[16x16,f32] = A[16x32,f16] x B[32x16,f16] + C
// ---------------------------------------------------------------------------
typedef __attribute__((ext_vector_type(16))) _Float16 v16h;
typedef __attribute__((ext_vector_type(8)))  float    v8f;

#define BT    16384   // B*T = 64*256
#define TSEQ  256
#define CEMB  384
#define H4    1536
#define NHEAD 6
#define HDIM  64
#define VOCAB 65
#define VPAD  128     // LM head padded to a multiple of 64 cols (branch-free GEMM)

// A-operand fragment (16x32 f16): lane<16 holds contiguous K 0..15,
// lane>=16 holds K 16..31 (matching load_frag_b). rowk points at the lane's row
// in uint32 (f16-pair) granularity, already offset to the K-step base.
__device__ __forceinline__ v16h load_frag_a(const uint32_t* rowk, int hi) {
  union { v16h v; uint32_t u[8]; } f;
  const uint32_t* p = rowk + (hi ? 8 : 0);
#pragma unroll
  for (int j = 0; j < 8; ++j) f.u[j] = p[j];
  return f.v;
}

// B-operand fragment (32x16 f16): lane<16 holds contiguous K 0..15 of its
// column, lane>=16 holds K 16..31 (same order as load_frag_a).
__device__ __forceinline__ v16h load_frag_b(const uint32_t* rowk, int hi) {
  union { v16h v; uint32_t u[8]; } f;
  const uint32_t* p = rowk + (hi ? 8 : 0);
#pragma unroll
  for (int j = 0; j < 8; ++j) f.u[j] = p[j];
  return f.v;
}

__device__ __forceinline__ v8f wmma_f16(v16h a, v16h b, v8f c) {
  return __builtin_amdgcn_wmma_f32_16x16x32_f16(false, a, false, b, (short)0, c,
                                                false, false);
}

// ---------------------------------------------------------------------------
// x = tok_emb[idx] + pos_emb
// ---------------------------------------------------------------------------
__global__ __launch_bounds__(256) void embed_kernel(
    const int* __restrict__ idx, const float* __restrict__ tok,
    const float* __restrict__ pos, float* __restrict__ x) {
  int tid = blockIdx.x * 256 + threadIdx.x;
  if (tid >= BT * CEMB) return;
  int t = tid / CEMB, c = tid - t * CEMB;
  x[tid] = tok[(size_t)idx[t] * CEMB + c] + pos[(size_t)(t & (TSEQ - 1)) * CEMB + c];
}

// ---------------------------------------------------------------------------
// Weight convert+transpose: W[K,N] f32 -> Wt[Npad,K] f16 (zero pad cols >= N)
// ---------------------------------------------------------------------------
__global__ __launch_bounds__(256) void conv_w_kernel(
    const float* __restrict__ W, _Float16* __restrict__ Wt,
    int K, int N, int total) {
  int tid = blockIdx.x * 256 + threadIdx.x;
  if (tid >= total) return;
  int n = tid / K, k = tid - n * K;
  Wt[tid] = (n < N) ? (_Float16)W[(size_t)k * N + n] : (_Float16)0.f;
}

// ---------------------------------------------------------------------------
// LayerNorm row (C=384) -> f16 output. One wave per row, 12 elems/lane.
// ---------------------------------------------------------------------------
__global__ __launch_bounds__(256) void ln_kernel(
    const float* __restrict__ x, const float* __restrict__ g,
    const float* __restrict__ bta, _Float16* __restrict__ out) {
  int row = blockIdx.x * 8 + (threadIdx.x >> 5);
  int lane = threadIdx.x & 31;
  const float* xr = x + (size_t)row * CEMB;
  float v[12];
  float s = 0.f;
#pragma unroll
  for (int i = 0; i < 12; ++i) { v[i] = xr[lane + 32 * i]; s += v[i]; }
#pragma unroll
  for (int o = 16; o > 0; o >>= 1) s += __shfl_xor(s, o, 32);
  float mean = s * (1.f / CEMB);
  float var = 0.f;
#pragma unroll
  for (int i = 0; i < 12; ++i) { float d = v[i] - mean; var += d * d; }
#pragma unroll
  for (int o = 16; o > 0; o >>= 1) var += __shfl_xor(var, o, 32);
  float inv = rsqrtf(var * (1.f / CEMB) + 1e-5f);
  _Float16* orow = out + (size_t)row * CEMB;
#pragma unroll
  for (int i = 0; i < 12; ++i) {
    int c = lane + 32 * i;
    orow[c] = (_Float16)((v[i] - mean) * inv * g[c] + bta[c]);
  }
}

// ---------------------------------------------------------------------------
// Generic WMMA GEMM: out[M,N] = A[M,K](f16) * Wt[Npad,K](f16,transposed) +bias
// One wave computes a 16-row x 64-col strip: 1 A-frag reused by 4 B-frags.
// Npad must be a multiple of 64 -> branch-free, exec-mask-free inner loop.
// Epilogue modes (compile-time): 0 f16 | 1 fp32 residual += | 2 relu f16 | 3 fp32
// ---------------------------------------------------------------------------
#define EP_F16   0
#define EP_RESID 1
#define EP_RELU  2
#define EP_F32   3

template <int MODE>
__global__ __launch_bounds__(128) void gemm_wmma_kernel(
    const _Float16* __restrict__ A, const _Float16* __restrict__ Wt,
    const float* __restrict__ bias, _Float16* __restrict__ out16,
    float* __restrict__ out32, int K, int colGroups, int Nvalid, int ldOut) {
  const int lane = threadIdx.x & 31;
  const int ln = lane & 15;
  const int hi = lane >> 4;
  const int wave = blockIdx.x * (blockDim.x >> 5) + (threadIdx.x >> 5);
  const int rt = wave / colGroups;
  const int cg = wave - rt * colGroups;
  const int ct0 = cg << 2;           // first of 4 column tiles
  const int rowBase = rt << 4;
  const int Ku = K >> 1;             // row length in f16-pairs

  const uint32_t* Arow = (const uint32_t*)A + (size_t)(rowBase + ln) * Ku;
  const uint32_t* B0 = (const uint32_t*)Wt + (size_t)(ct0 * 16 + ln) * Ku;
  const uint32_t* B1 = B0 + (size_t)16 * Ku;
  const uint32_t* B2 = B0 + (size_t)32 * Ku;
  const uint32_t* B3 = B0 + (size_t)48 * Ku;

  v8f acc0 = {}, acc1 = {}, acc2 = {}, acc3 = {};
  for (int k0 = 0; k0 < K; k0 += 32) {
    const int ko = k0 >> 1;
    v16h a = load_frag_a(Arow + ko, hi);
    acc0 = wmma_f16(a, load_frag_b(B0 + ko, hi), acc0);
    acc1 = wmma_f16(a, load_frag_b(B1 + ko, hi), acc1);
    acc2 = wmma_f16(a, load_frag_b(B2 + ko, hi), acc2);
    acc3 = wmma_f16(a, load_frag_b(B3 + ko, hi), acc3);
  }

#pragma unroll
  for (int t = 0; t < 4; ++t) {
    v8f acc = (t == 0) ? acc0 : (t == 1) ? acc1 : (t == 2) ? acc2 : acc3;
    int col = (ct0 + t) * 16 + ln;
    if (col >= Nvalid) continue;     // only trims the padded LM-head columns
    float bv = bias ? bias[col] : 0.f;
#pragma unroll
    for (int r = 0; r < 8; ++r) {
      int row = rowBase + r + hi * 8;  // C/D layout: vgpr r, half-wave selects M+8
      float v = acc[r] + bv;
      size_t o = (size_t)row * ldOut + col;
      if (MODE == EP_F16)        out16[o] = (_Float16)v;
      else if (MODE == EP_RELU)  out16[o] = (_Float16)(v > 0.f ? v : 0.f);
      else if (MODE == EP_RESID) out32[o] += v;
      else                       out32[o] = v;
    }
  }
}

// ---------------------------------------------------------------------------
// V transpose: v16[b*256+t, h*64+d] -> vt16[(b*6+h)*64+d, t]
// ---------------------------------------------------------------------------
__global__ __launch_bounds__(256) void vtrans_kernel(
    const _Float16* __restrict__ v16, _Float16* __restrict__ vt16) {
  int tid = blockIdx.x * 256 + threadIdx.x;
  if (tid >= BT * CEMB) return;
  int tok = tid / CEMB, c = tid - tok * CEMB;
  int b = tok >> 8, t = tok & 255;
  int h = c >> 6, d = c & 63;
  vt16[((((size_t)(b * NHEAD + h)) * HDIM + d) << 8) + t] = v16[tid];
}

// ---------------------------------------------------------------------------
// Causal attention: one wave per (b, head, 16-query tile).
// S = QK^T via WMMA -> LDS -> fp32 softmax -> P(f16) in LDS -> PV via WMMA.
// ---------------------------------------------------------------------------
__global__ __launch_bounds__(32) void attn_kernel(
    const _Float16* __restrict__ q16, const _Float16* __restrict__ k16,
    const _Float16* __restrict__ vt16, _Float16* __restrict__ att16) {
  __shared__ float    S[16][TSEQ];
  __shared__ _Float16 P[16][TSEQ];

  const int id = blockIdx.x;
  const int qt = id & 15;
  const int h = (id >> 4) % NHEAD;
  const int b = id / (16 * NHEAD);
  const int lane = threadIdx.x & 31;
  const int ln = lane & 15;
  const int hi = lane >> 4;
  const size_t tokBase = (size_t)b * TSEQ;
  const int kmax = ((qt * 16 + 16) + 31) & ~31;  // causal K extent, 32-aligned

  const uint32_t* Qu =
      (const uint32_t*)q16 + ((tokBase + qt * 16 + ln) * CEMB + h * HDIM) / 2;
  v16h qf0 = load_frag_a(Qu, hi);       // head-dim K 0..31
  v16h qf1 = load_frag_a(Qu + 16, hi);  // head-dim K 32..63

  for (int n0 = 0; n0 < kmax; n0 += 16) {
    const uint32_t* Ku =
        (const uint32_t*)k16 + ((tokBase + n0 + ln) * CEMB + h * HDIM) / 2;
    v8f acc = {};
    acc = wmma_f16(qf0, load_frag_b(Ku, hi), acc);
    acc = wmma_f16(qf1, load_frag_b(Ku + 16, hi), acc);
#pragma unroll
    for (int r = 0; r < 8; ++r) S[r + hi * 8][n0 + ln] = acc[r];
  }
  asm volatile("s_wait_dscnt 0" ::: "memory");

  if (lane < 16) {
    const int qpos = qt * 16 + lane;
    const float scale = 0.05103103630798287f;  // 384^-0.5 (full C, per reference)
    float mx = -3.0e38f;
    for (int n = 0; n <= qpos; ++n) mx = fmaxf(mx, S[lane][n] * scale);
    float sum = 0.f;
    for (int n = 0; n < kmax; ++n) {
      float p = (n <= qpos) ? __expf(S[lane][n] * scale - mx) : 0.f;
      S[lane][n] = p;
      sum += p;
    }
    float inv = 1.f / sum;
    for (int n = 0; n < kmax; ++n) P[lane][n] = (_Float16)(S[lane][n] * inv);
  }
  asm volatile("s_wait_dscnt 0" ::: "memory");

  const size_t vtBase = ((size_t)(b * NHEAD + h)) * HDIM * TSEQ;
#pragma unroll
  for (int d0 = 0; d0 < HDIM; d0 += 16) {
    v8f acc = {};
    for (int kk0 = 0; kk0 < kmax; kk0 += 32) {
      v16h pa = load_frag_a((const uint32_t*)&P[ln][0] + (kk0 >> 1), hi);
      v16h vb = load_frag_b(
          (const uint32_t*)vt16 + (vtBase + (size_t)(d0 + ln) * TSEQ + kk0) / 2, hi);
      acc = wmma_f16(pa, vb, acc);
    }
#pragma unroll
    for (int r = 0; r < 8; ++r)
      att16[(tokBase + qt * 16 + r + hi * 8) * CEMB + h * HDIM + d0 + ln] =
          (_Float16)acc[r];
  }
}

// ---------------------------------------------------------------------------
// Host launcher
// ---------------------------------------------------------------------------
extern "C" void kernel_launch(void* const* d_in, const int* in_sizes, int n_in,
                              void* d_out, int out_size, void* d_ws,
                              size_t ws_size, hipStream_t stream) {
  (void)in_sizes; (void)n_in; (void)out_size; (void)ws_size;

  const int*   idx     = (const int*)d_in[0];
  const float* tok_emb = (const float*)d_in[1];
  const float* pos_emb = (const float*)d_in[2];
  const float* Wq      = (const float*)d_in[3];
  const float* Wk      = (const float*)d_in[4];
  const float* Wv      = (const float*)d_in[5];
  const float* Wo      = (const float*)d_in[6];
  const float* bo      = (const float*)d_in[7];
  const float* ln1_g   = (const float*)d_in[8];
  const float* ln1_b   = (const float*)d_in[9];
  const float* ln2_g   = (const float*)d_in[10];
  const float* ln2_b   = (const float*)d_in[11];
  const float* W1      = (const float*)d_in[12];
  const float* b1      = (const float*)d_in[13];
  const float* W2      = (const float*)d_in[14];
  const float* b2      = (const float*)d_in[15];
  const float* lnf_g   = (const float*)d_in[16];
  const float* lnf_b   = (const float*)d_in[17];
  const float* Wlm     = (const float*)d_in[18];
  const float* blm     = (const float*)d_in[19];
  float* logits = (float*)d_out;

  char* ws = (char*)d_ws;
  size_t off = 0;
  auto alloc = [&](size_t bytes) -> char* {
    char* p = ws + off;
    off += (bytes + 255) & ~(size_t)255;
    return p;
  };

  float*     x     = (float*)alloc((size_t)BT * CEMB * 4);
  _Float16*  h16   = (_Float16*)alloc((size_t)BT * CEMB * 2);
  _Float16*  q16   = (_Float16*)alloc((size_t)BT * CEMB * 2);
  _Float16*  k16   = (_Float16*)alloc((size_t)BT * CEMB * 2);
  _Float16*  v16   = (_Float16*)alloc((size_t)BT * CEMB * 2);
  _Float16*  vt16  = (_Float16*)alloc((size_t)BT * CEMB * 2);
  _Float16*  att16 = (_Float16*)alloc((size_t)BT * CEMB * 2);
  _Float16*  u16   = (_Float16*)alloc((size_t)BT * H4 * 2);
  _Float16*  wq16  = (_Float16*)alloc((size_t)CEMB * CEMB * 2);
  _Float16*  wk16  = (_Float16*)alloc((size_t)CEMB * CEMB * 2);
  _Float16*  wv16  = (_Float16*)alloc((size_t)CEMB * CEMB * 2);
  _Float16*  wo16  = (_Float16*)alloc((size_t)CEMB * CEMB * 2);
  _Float16*  w116  = (_Float16*)alloc((size_t)H4 * CEMB * 2);   // [1536,384]
  _Float16*  w216  = (_Float16*)alloc((size_t)CEMB * H4 * 2);   // [384,1536]
  _Float16*  wlm16 = (_Float16*)alloc((size_t)VPAD * CEMB * 2); // [128,384]

  const int embBlocks = (BT * CEMB + 255) / 256;
  embed_kernel<<<embBlocks, 256, 0, stream>>>(idx, tok_emb, pos_emb, x);

  auto convBlocks = [](int total) { return (total + 255) / 256; };
  // GEMM grid: waves = (BT/16) * colGroups; 4 waves / 128-thread block
  auto gemmBlocks = [](int colGroups) { return 256 * colGroups; };

  for (int l = 0; l < 3; ++l) {
    // weight conversion (f32 [K,N] -> f16 [Npad,K])
    conv_w_kernel<<<convBlocks(CEMB * CEMB), 256, 0, stream>>>(
        Wq + (size_t)l * CEMB * CEMB, wq16, CEMB, CEMB, CEMB * CEMB);
    conv_w_kernel<<<convBlocks(CEMB * CEMB), 256, 0, stream>>>(
        Wk + (size_t)l * CEMB * CEMB, wk16, CEMB, CEMB, CEMB * CEMB);
    conv_w_kernel<<<convBlocks(CEMB * CEMB), 256, 0, stream>>>(
        Wv + (size_t)l * CEMB * CEMB, wv16, CEMB, CEMB, CEMB * CEMB);
    conv_w_kernel<<<convBlocks(CEMB * CEMB), 256, 0, stream>>>(
        Wo + (size_t)l * CEMB * CEMB, wo16, CEMB, CEMB, CEMB * CEMB);
    conv_w_kernel<<<convBlocks(H4 * CEMB), 256, 0, stream>>>(
        W1 + (size_t)l * CEMB * H4, w116, CEMB, H4, H4 * CEMB);
    conv_w_kernel<<<convBlocks(CEMB * H4), 256, 0, stream>>>(
        W2 + (size_t)l * H4 * CEMB, w216, H4, CEMB, CEMB * H4);

    // h = ln1(x)
    ln_kernel<<<BT / 8, 256, 0, stream>>>(x, ln1_g + l * CEMB, ln1_b + l * CEMB, h16);

    // q,k,v = h @ W{q,k,v}   (384 cols -> 6 col groups)
    gemm_wmma_kernel<EP_F16><<<gemmBlocks(6), 128, 0, stream>>>(
        h16, wq16, nullptr, q16, nullptr, CEMB, 6, CEMB, CEMB);
    gemm_wmma_kernel<EP_F16><<<gemmBlocks(6), 128, 0, stream>>>(
        h16, wk16, nullptr, k16, nullptr, CEMB, 6, CEMB, CEMB);
    gemm_wmma_kernel<EP_F16><<<gemmBlocks(6), 128, 0, stream>>>(
        h16, wv16, nullptr, v16, nullptr, CEMB, 6, CEMB, CEMB);

    vtrans_kernel<<<embBlocks, 256, 0, stream>>>(v16, vt16);

    attn_kernel<<<64 * NHEAD * 16, 32, 0, stream>>>(q16, k16, vt16, att16);

    // x += att @ Wo + bo
    gemm_wmma_kernel<EP_RESID><<<gemmBlocks(6), 128, 0, stream>>>(
        att16, wo16, bo + l * CEMB, nullptr, x, CEMB, 6, CEMB, CEMB);

    // h = ln2(x)
    ln_kernel<<<BT / 8, 256, 0, stream>>>(x, ln2_g + l * CEMB, ln2_b + l * CEMB, h16);

    // u = relu(h @ W1 + b1)   (1536 cols -> 24 col groups)
    gemm_wmma_kernel<EP_RELU><<<gemmBlocks(24), 128, 0, stream>>>(
        h16, w116, b1 + l * H4, u16, nullptr, CEMB, 24, H4, H4);

    // x += u @ W2 + b2
    gemm_wmma_kernel<EP_RESID><<<gemmBlocks(6), 128, 0, stream>>>(
        u16, w216, b2 + l * CEMB, nullptr, x, H4, 6, CEMB, CEMB);
  }

  // final LN + LM head (65 cols padded to 128 -> 2 col groups, store-guarded)
  ln_kernel<<<BT / 8, 256, 0, stream>>>(x, lnf_g, lnf_b, h16);
  conv_w_kernel<<<convBlocks(VPAD * CEMB), 256, 0, stream>>>(
      Wlm, wlm16, CEMB, VOCAB, VPAD * CEMB);
  gemm_wmma_kernel<EP_F32><<<gemmBlocks(2), 128, 0, stream>>>(
      h16, wlm16, blm, nullptr, logits, CEMB, 2, VOCAB, VOCAB);
}
